// pLayer_14912126452420
// MI455X (gfx1250) — matrix operand order repaired and checked by reading, not compiled, not run on hardware
//
#include <hip/hip_runtime.h>
#include <math.h>

typedef __attribute__((ext_vector_type(16))) _Float16 v16h;
typedef __attribute__((ext_vector_type(8)))  float    v8f;

#define E_TOT   131072
#define N_IN    256
#define MDIM    258     // n_in + 2
#define MPAD    288     // 9 * 32 (K padding for WMMA)
#define NDIM    128
#define BM      128     // E-rows per block (8 waves * 16)
#define GMIN_C  0.01f
#define GMAX_C  10.0f

// Hardware tanh (V_TANH_F32 trans op) if the toolchain exposes it for gfx1250.
#if defined(__gfx1250__) && __has_builtin(__builtin_amdgcn_tanhf)
#define TANHF(x) __builtin_amdgcn_tanhf(x)
#else
#define TANHF(x) tanhf(x)
#endif

__device__ __forceinline__ float inv_f(float a) {   // INV: Q=(0.9,0.5,0.2,5)
    return 0.9f - 0.5f * TANHF((a - 0.2f) * 5.0f);
}
__device__ __forceinline__ float act_f(float z) {   // ACT: ETA=(0.1,0.5,0.2,5)
    return 0.1f + 0.5f * TANHF((z - 0.2f) * 5.0f);
}
__device__ __forceinline__ float clamp_prune(float t) {
    float tc = fminf(fmaxf(t, -GMAX_C), GMAX_C);
    return (fabsf(tc) < GMIN_C) ? 0.0f : tc;
}

// ---------------- prep kernel 1: per-column (n) weight/power matrices -------
// Writes transposed, K-padded f16 operand matrices [NDIM][MPAD] so that the
// WMMA B-layout load is 16 contiguous halves per lane.
__global__ void prep_cols(const float* __restrict__ theta,
                          _Float16* __restrict__ WpT, _Float16* __restrict__ WnT,
                          _Float16* __restrict__ GpT, _Float16* __restrict__ GnT,
                          float* __restrict__ Gs, float* __restrict__ colScale) {
    const int n = threadIdx.x;            // 128 threads, one column each
    float S = 0.0f, mn = 3.4e38f;
    for (int m = 0; m < MDIM; ++m) {
        float t = theta[m * NDIM + n];
        mn = fminf(mn, fabsf(t));         // g_init = |theta_raw|
        S += fabsf(clamp_prune(t));
    }
    const float scale = 0.1f / mn;        // PGMIN / min_col(g_init)
    const float invS  = 1.0f / S;
    float gsum = 0.0f;
    for (int m = 0; m < MDIM; ++m) {
        float t   = theta[m * NDIM + n];
        float eff = clamp_prune(t);
        float w   = fabsf(eff) * invS;
        bool  pos = (eff >= 0.0f);
        float gt  = fabsf(t) * scale;     // g_tilde
        gsum += gt;
        size_t idx = (size_t)n * MPAD + m;
        WpT[idx] = pos ? (_Float16)w  : (_Float16)0.0f;
        WnT[idx] = pos ? (_Float16)0.0f : (_Float16)w;
        GpT[idx] = pos ? (_Float16)gt : (_Float16)0.0f;
        GnT[idx] = pos ? (_Float16)0.0f : (_Float16)gt;
    }
    for (int m = MDIM; m < MPAD; ++m) {   // zero K-padding
        size_t idx = (size_t)n * MPAD + m;
        WpT[idx] = (_Float16)0.0f; WnT[idx] = (_Float16)0.0f;
        GpT[idx] = (_Float16)0.0f; GnT[idx] = (_Float16)0.0f;
    }
    Gs[n]       = gsum;                   // sum_m g_tilde  (for Sy2 term)
    colScale[n] = scale;
}

// ---------------- prep kernel 2: per-row (m) g_tilde row sums ---------------
__global__ void prep_rows(const float* __restrict__ theta,
                          const float* __restrict__ colScale,
                          float* __restrict__ Gp, float* __restrict__ Gn) {
    const int m = threadIdx.x;            // 288 threads
    if (m >= MPAD) return;
    if (m >= MDIM) { Gp[m] = 0.0f; Gn[m] = 0.0f; return; }
    float gp = 0.0f, gn = 0.0f;
    for (int n = 0; n < NDIM; ++n) {
        float t   = theta[m * NDIM + n];
        float eff = clamp_prune(t);
        float gt  = fabsf(t) * colScale[n];
        if (eff >= 0.0f) gp += gt; else gn += gt;
    }
    Gp[m] = gp; Gn[m] = gn;
}

// ---------------- main fused kernel -----------------------------------------
// z  = x@Wp + xn@Wn              (WMMA, f16 in / f32 acc)
// uv = x@(g~pos) + xn@(g~neg)    (same A operands, 2 extra WMMA passes)
// a_new = ACT(z);  mac_power contributions:
//   sX = sum x^2*Gp[m] + xn^2*Gn[m];  cr = sum Gs[n]*z^2 - 2*z*uv
//
// Each of the 4 passes below issues 8 WMMAs on *distinct* accumulators, so
// dependent WMMAs on the same accumulator are 8 instructions apart -> no
// F16-WMMA RAW hazard NOPs (ISA 7.12.1 needs dist >= 5).
__device__ __forceinline__ void wmma_pass(const v16h& a, int bko, int l15,
                                          const _Float16* __restrict__ B,
                                          v8f acc[8]) {
#pragma unroll
    for (int nt = 0; nt < 8; ++nt) {
        const size_t boff = (size_t)(nt * 16 + l15) * MPAD + bko;
        v16h b = *(const v16h*)(B + boff);
        acc[nt] = __builtin_amdgcn_wmma_f32_16x16x32_f16(false, a, false, b,
                                                         (short)0, acc[nt], false, false);
    }
}

__global__ __launch_bounds__(256, 1)
void player_main(const float* __restrict__ x,
                 const _Float16* __restrict__ WpT, const _Float16* __restrict__ WnT,
                 const _Float16* __restrict__ GpT, const _Float16* __restrict__ GnT,
                 const float* __restrict__ Gp, const float* __restrict__ Gn,
                 const float* __restrict__ Gs,
                 float* __restrict__ out, float* __restrict__ macp) {
    __shared__ float sGp[MPAD], sGn[MPAD], sGs[NDIM];
    __shared__ float red[256];
    const int tid  = threadIdx.x;
    const int wave = tid >> 5;
    const int lane = tid & 31;
    const int l15  = lane & 15;
    const int hi   = lane >> 4;

    for (int i = tid; i < MPAD; i += 256) { sGp[i] = Gp[i]; sGn[i] = Gn[i]; }
    if (tid < NDIM) sGs[tid] = Gs[tid];
    __syncthreads();

    // A-matrix 16x32 f16 layout: this lane holds row M=l15; half h maps to
    // K = (hi*16) + h.
    const int row = blockIdx.x * BM + wave * 16 + l15;

    v8f zero = {};
    v8f accZ[8], accU[8];
#pragma unroll
    for (int i = 0; i < 8; ++i) { accZ[i] = zero; accU[i] = zero; }

    float sX = 0.0f;

    // ---- k-steps 0..255: real a_previous columns ----
    for (int k0 = 0; k0 < N_IN; k0 += 32) {
        const int kb = k0 + (hi << 4);
        const float* xr = x + (size_t)row * N_IN + kb;
        const float4 f0 = *(const float4*)(xr);
        const float4 f1 = *(const float4*)(xr + 4);
        const float4 f2 = *(const float4*)(xr + 8);
        const float4 f3 = *(const float4*)(xr + 12);
        float vals[16] = { f0.x, f0.y, f0.z, f0.w, f1.x, f1.y, f1.z, f1.w,
                           f2.x, f2.y, f2.z, f2.w, f3.x, f3.y, f3.z, f3.w };
        v16h ax, an;
#pragma unroll
        for (int h = 0; h < 16; ++h) {
            const int col = kb + h;
            const float v  = vals[h];
            const float vn = inv_f(v);                      // cols < 256 always
            sX += v * v * sGp[col] + vn * vn * sGn[col];
            ax[h] = (_Float16)v;
            an[h] = (_Float16)vn;
        }
        const int bko = k0 + (hi << 4);
        wmma_pass(ax, bko, l15, WpT, accZ);
        wmma_pass(an, bko, l15, WnT, accZ);
        wmma_pass(ax, bko, l15, GpT, accU);
        wmma_pass(an, bko, l15, GnT, accU);
    }

    // ---- tail k-step 256..287: ones col (256), zeros col (257), padding ----
    {
        const int kb = N_IN + (hi << 4);
        v16h ax, an;
#pragma unroll
        for (int h = 0; h < 16; ++h) {
            const int col = kb + h;
            const float v  = (col == N_IN) ? 1.0f : 0.0f;   // x: [.., 1, 0, pad]
            const float vn = (col == N_IN) ? inv_f(1.0f) : 0.0f; // xn last col forced 0
            sX += v * v * sGp[col] + vn * vn * sGn[col];
            ax[h] = (_Float16)v;
            an[h] = (_Float16)vn;
        }
        const int bko = N_IN + (hi << 4);
        wmma_pass(ax, bko, l15, WpT, accZ);
        wmma_pass(an, bko, l15, WnT, accZ);
        wmma_pass(ax, bko, l15, GpT, accU);
        wmma_pass(an, bko, l15, GnT, accU);
    }

    // ---- epilogue: activation store + power terms ----
    // D layout: VGPR j, lanes 0-15 -> M=j, lanes 16-31 -> M=8+j; N = lane%16.
    float cr = 0.0f;
    const int orow = blockIdx.x * BM + wave * 16 + hi * 8;
#pragma unroll
    for (int nt = 0; nt < 8; ++nt) {
        const int ncol = nt * 16 + l15;
        const float gs = sGs[ncol];
#pragma unroll
        for (int j = 0; j < 8; ++j) {
            const float zv = accZ[nt][j];
            const float uv = accU[nt][j];
            cr += zv * (gs * zv - 2.0f * uv);
            out[(size_t)(orow + j) * NDIM + ncol] = act_f(zv);
        }
    }

    red[tid] = sX + cr;
    __syncthreads();
    for (int s = 128; s > 0; s >>= 1) {
        if (tid < s) red[tid] += red[tid + s];
        __syncthreads();
    }
    if (tid == 0) atomicAdd(macp, red[0] * (1.0f / (float)E_TOT));
}

// ---------------- host side -------------------------------------------------
extern "C" void kernel_launch(void* const* d_in, const int* in_sizes, int n_in,
                              void* d_out, int out_size, void* d_ws, size_t ws_size,
                              hipStream_t stream) {
    const float* a_prev = (const float*)d_in[0];   // [E, 256]
    const float* theta  = (const float*)d_in[1];   // [258, 128]
    float* out = (float*)d_out;                    // [E*128] a_new + [1] mac_power

    char* ws = (char*)d_ws;
    const size_t MAT = (size_t)NDIM * MPAD * sizeof(_Float16);  // 73728 B
    _Float16* WpT = (_Float16*)(ws + 0 * MAT);
    _Float16* WnT = (_Float16*)(ws + 1 * MAT);
    _Float16* GpT = (_Float16*)(ws + 2 * MAT);
    _Float16* GnT = (_Float16*)(ws + 3 * MAT);
    float* fws      = (float*)(ws + 4 * MAT);
    float* Gs       = fws;            // 128
    float* colScale = fws + NDIM;     // 128
    float* Gp       = fws + 2 * NDIM;         // 288
    float* Gn       = fws + 2 * NDIM + MPAD;  // 288

    float* macp = out + (size_t)E_TOT * NDIM;
    hipMemsetAsync(macp, 0, sizeof(float), stream);

    prep_cols<<<1, NDIM, 0, stream>>>(theta, WpT, WnT, GpT, GnT, Gs, colScale);
    prep_rows<<<1, MPAD, 0, stream>>>(theta, colScale, Gp, Gn);
    player_main<<<E_TOT / BM, 256, 0, stream>>>(a_prev, WpT, WnT, GpT, GnT,
                                                Gp, Gn, Gs, out, macp);
}
